// SwinBlock1D_39522289058394
// MI455X (gfx1250) — matrix operand order repaired and checked by reading, not compiled, not run on hardware
//
#include <hip/hip_runtime.h>
#include <hip/hip_bf16.h>

typedef __attribute__((ext_vector_type(16))) __bf16 v16bf;
typedef __attribute__((ext_vector_type(8)))  __bf16 v8bf;
typedef __attribute__((ext_vector_type(8)))  float  v8f;

#define TOKENS 65536
#define CDIM   512
#define NSEQ   8192
#define SHIFTN 4

#define BM 128
#define BN 256
#define BK 32
#define LDT 40   // padded LDS stride (bf16 elems) -> conflict-free frag reads

constexpr int EPI_QKV  = 0;
constexpr int EPI_PROJ = 1;
constexpr int EPI_GELU = 2;
constexpr int EPI_MLP2 = 3;

// ---------------- weight convert + transpose: W(K,N) f32 -> Wt(N,K) bf16 ----
__global__ __launch_bounds__(256) void wt_convert(const float* __restrict__ W,
        __hip_bfloat16* __restrict__ Wt, int K, int N, long total)
{
    long i = (long)blockIdx.x * 256 + threadIdx.x;
    if (i >= total) return;
    int k = (int)(i / N), n = (int)(i % N);
    Wt[(long)n * K + k] = __float2bfloat16(W[i]);
}

// ---------------- LayerNorm (one token per block) + optional roll(-SHIFT) ----
__global__ __launch_bounds__(256) void ln_bf16(const float* __restrict__ x,
    const float* __restrict__ g, const float* __restrict__ bta,
    __hip_bfloat16* __restrict__ out, int doRoll)
{
    int t = blockIdx.x;
    int tid = threadIdx.x, lane = tid & 31, wid = tid >> 5;
    const float2 v = reinterpret_cast<const float2*>(x + (size_t)t * CDIM)[tid];
    float s = v.x + v.y;
    float q = v.x * v.x + v.y * v.y;
    #pragma unroll
    for (int o = 16; o > 0; o >>= 1) {
        s += __shfl_xor(s, o, 32);
        q += __shfl_xor(q, o, 32);
    }
    __shared__ float ps[8], pq[8];
    if (lane == 0) { ps[wid] = s; pq[wid] = q; }
    __syncthreads();
    float S = 0.f, Q = 0.f;
    #pragma unroll
    for (int i = 0; i < 8; ++i) { S += ps[i]; Q += pq[i]; }
    float mean = S * (1.f / CDIM);
    float var  = Q * (1.f / CDIM) - mean * mean;
    float rs   = rsqrtf(var + 1e-5f);
    int td = t;
    if (doRoll) {
        int b = t >> 13, n = t & (NSEQ - 1);
        td = (b << 13) | ((n - SHIFTN) & (NSEQ - 1));   // h[(n-4)%N] = LN(x)[n]
    }
    int c = tid * 2;
    __hip_bfloat16* orow = out + (size_t)td * CDIM;
    orow[c]     = __float2bfloat16((v.x - mean) * rs * g[c]     + bta[c]);
    orow[c + 1] = __float2bfloat16((v.y - mean) * rs * g[c + 1] + bta[c + 1]);
}

// ---------------- WMMA fragment loads from padded row-major LDS tiles -------
// 16-bit A 16x32: lane = M row (lane&15); lanes<16 -> K 0..7 & 16..23,
// lanes>=16 -> K 8..15 & 24..31 (ISA 7.12.2).
__device__ __forceinline__ v16bf frag_a(const __hip_bfloat16* tile, int lane) {
    int m  = lane & 15;
    int kk = (lane >> 4) << 3;
    const __bf16* p = reinterpret_cast<const __bf16*>(tile + m * LDT + kk);
    v8bf lo = *reinterpret_cast<const v8bf*>(p);
    v8bf hi = *reinterpret_cast<const v8bf*>(p + 16);
    v16bf r;
    #pragma unroll
    for (int i = 0; i < 8; ++i) { r[i] = lo[i]; r[i + 8] = hi[i]; }
    return r;
}
// 16-bit B 32x16 (from Wt tile, rows = N): lane = N col; lanes<16 hold K 0..7 & 16..23,
// lanes>=16 hold K 8..15 & 24..31.
__device__ __forceinline__ v16bf frag_b(const __hip_bfloat16* tile, int lane) {
    int n  = lane & 15;
    int kk = (lane >> 4) << 3;
    const __bf16* p = reinterpret_cast<const __bf16*>(tile + n * LDT + kk);
    v8bf lo = *reinterpret_cast<const v8bf*>(p);
    v8bf hi = *reinterpret_cast<const v8bf*>(p + 16);
    v16bf r;
    #pragma unroll
    for (int i = 0; i < 8; ++i) { r[i] = lo[i]; r[i + 8] = hi[i]; }
    return r;
}

// ------- tiled bf16 WMMA GEMM, double-buffered LDS: D = A @ Wt^T + epilogue --
// 256 thr = 8 waves as 2(M) x 4(N); each wave owns a 64x64 tile (4x4 WMMA).
template <int EPI>
__global__ __launch_bounds__(256) void gemm_wmma(
    const __hip_bfloat16* __restrict__ A,
    const __hip_bfloat16* __restrict__ Wt,
    const float* __restrict__ bias,
    const float* __restrict__ resid,   // PROJ: x (read at rolled row)
    float* outF,                       // PROJ: x2; MLP2: final out (RMW)
    __hip_bfloat16* __restrict__ outH, // QKV / GELU bf16 out
    int N, int K)
{
    __shared__ __hip_bfloat16 As[2][BM * LDT];   // 2 x 10 KB
    __shared__ __hip_bfloat16 Bs[2][BN * LDT];   // 2 x 20 KB
    const int tid  = threadIdx.x;
    const int lane = tid & 31, wid = tid >> 5;
    const long mBase = (long)blockIdx.y * BM;
    const int  nBase = blockIdx.x * BN;
    const int wm = (wid & 1) << 6;   // wave M offset: 0/64
    const int wn = (wid >> 1) << 6;  // wave N offset: 0/64/128/192

    v8f acc[4][4];
    #pragma unroll
    for (int i = 0; i < 4; ++i)
        #pragma unroll
        for (int j = 0; j < 4; ++j)
            #pragma unroll
            for (int e = 0; e < 8; ++e) acc[i][j][e] = 0.f;

    const int lr = tid >> 2;        // 0..63
    const int lk = (tid & 3) << 3;  // 0,8,16,24
    const __hip_bfloat16* Abase = A  + (mBase + lr) * K + lk;
    const __hip_bfloat16* Bbase = Wt + (long)(nBase + lr) * K + lk;

    uint4 ra0, ra1, rb0, rb1, rb2, rb3;
    auto gload = [&](int k0) {
        ra0 = *reinterpret_cast<const uint4*>(Abase + k0);
        ra1 = *reinterpret_cast<const uint4*>(Abase + (long)64 * K + k0);
        rb0 = *reinterpret_cast<const uint4*>(Bbase + k0);
        rb1 = *reinterpret_cast<const uint4*>(Bbase + (long)64  * K + k0);
        rb2 = *reinterpret_cast<const uint4*>(Bbase + (long)128 * K + k0);
        rb3 = *reinterpret_cast<const uint4*>(Bbase + (long)192 * K + k0);
    };
    auto sstore = [&](int buf) {
        *reinterpret_cast<uint4*>(&As[buf][lr * LDT + lk])         = ra0;
        *reinterpret_cast<uint4*>(&As[buf][(lr + 64) * LDT + lk])  = ra1;
        *reinterpret_cast<uint4*>(&Bs[buf][lr * LDT + lk])         = rb0;
        *reinterpret_cast<uint4*>(&Bs[buf][(lr + 64) * LDT + lk])  = rb1;
        *reinterpret_cast<uint4*>(&Bs[buf][(lr + 128) * LDT + lk]) = rb2;
        *reinterpret_cast<uint4*>(&Bs[buf][(lr + 192) * LDT + lk]) = rb3;
    };

    gload(0);
    sstore(0);
    __syncthreads();

    const int nIter = K / BK;
    int cur = 0;
    for (int it = 0; it < nIter; ++it) {
        if (it + 1 < nIter) gload((it + 1) * BK);   // prefetch next K-slice
        v16bf bf[4];
        #pragma unroll
        for (int nt = 0; nt < 4; ++nt)
            bf[nt] = frag_b(&Bs[cur][(wn + nt * 16) * LDT], lane);
        #pragma unroll
        for (int mt = 0; mt < 4; ++mt) {
            v16bf af = frag_a(&As[cur][(wm + mt * 16) * LDT], lane);
            #pragma unroll
            for (int nt = 0; nt < 4; ++nt)
                acc[mt][nt] = __builtin_amdgcn_wmma_f32_16x16x32_bf16(
                    false, af, false, bf[nt], (short)0, acc[mt][nt], false, false);
        }
        if (it + 1 < nIter) sstore(cur ^ 1);        // fill other buffer
        __syncthreads();
        cur ^= 1;
    }

    // epilogue: VGPR r of a 16x16 f32 tile = row r (lanes 0-15) / r+8 (16-31)
    const int colLane = lane & 15;
    const int rowHi   = (lane >> 4) << 3;
    #pragma unroll
    for (int nt = 0; nt < 4; ++nt) {
        int col = nBase + wn + nt * 16 + colLane;
        float bval = bias[col];
        #pragma unroll
        for (int mt = 0; mt < 4; ++mt) {
            long rowBase = mBase + wm + mt * 16 + rowHi;
            #pragma unroll
            for (int r = 0; r < 8; ++r) {
                long row = rowBase + r;
                float v = acc[mt][nt][r] + bval;
                if (EPI == EPI_QKV) {
                    outH[row * N + col] = __float2bfloat16(v);
                } else if (EPI == EPI_GELU) {
                    float gl = 0.5f * v * (1.f + erff(v * 0.70710678118f));
                    outH[row * N + col] = __float2bfloat16(gl);
                } else if (EPI == EPI_PROJ) {
                    long b   = row >> 13;
                    long n   = row & (NSEQ - 1);
                    long dst = (b << 13) | ((n + SHIFTN) & (NSEQ - 1)); // roll(+4)
                    long idx = dst * CDIM + col;
                    outF[idx] = v + resid[idx];                          // + shortcut x
                } else { // EPI_MLP2
                    long idx = row * CDIM + col;
                    outF[idx] = v + outF[idx];                           // + x2 (in d_out)
                }
            }
        }
    }
}

// ---------------- window-8 attention: one wave per (window, head) ------------
__global__ __launch_bounds__(256) void attn_win8(
    const __hip_bfloat16* __restrict__ qkv,  // (T,1536): q|k|v, head-major cols
    __hip_bfloat16* __restrict__ o)          // (T,512)
{
    int wid = threadIdx.x >> 5, lane = threadIdx.x & 31;
    int task = blockIdx.x * 8 + wid;
    int win = task >> 3, h = task & 7;
    const __hip_bfloat16* base = qkv + (size_t)win * 8 * 1536 + h * 64;

    // scores: lane covers pairs p=2l,2l+1 -> same query row i, cols j0,j0+1
    int p = lane * 2;
    int i = p >> 3, j0 = p & 7;
    const __hip_bfloat16* qr  = base + (size_t)i  * 1536;
    const __hip_bfloat16* k0r = base + (size_t)j0 * 1536 + 512;
    const __hip_bfloat16* k1r = k0r + 1536;
    float s0 = 0.f, s1 = 0.f;
    #pragma unroll
    for (int d = 0; d < 64; ++d) {
        float qd = __bfloat162float(qr[d]);
        s0 += qd * __bfloat162float(k0r[d]);
        s1 += qd * __bfloat162float(k1r[d]);
    }
    s0 *= 0.125f; s1 *= 0.125f;           // hd^-0.5, hd=64
    // softmax over row i (8 scores spread across lanes 4i..4i+3)
    float m = fmaxf(s0, s1);
    m = fmaxf(m, __shfl_xor(m, 1, 32));
    m = fmaxf(m, __shfl_xor(m, 2, 32));
    float e0 = __expf(s0 - m), e1 = __expf(s1 - m);
    float ssum = e0 + e1;
    ssum += __shfl_xor(ssum, 1, 32);
    ssum += __shfl_xor(ssum, 2, 32);
    float inv = 1.f / ssum;
    float a0 = e0 * inv, a1 = e1 * inv;

    // output: lane -> row oi = lane/4, d-chunk c = lane%4 (16 dims each)
    int oi = lane >> 2, c = lane & 3;
    float oacc[16];
    #pragma unroll
    for (int d = 0; d < 16; ++d) oacc[d] = 0.f;
    #pragma unroll
    for (int j = 0; j < 8; ++j) {
        int src = (oi << 2) + (j >> 1);
        float aj = __shfl((j & 1) ? a1 : a0, src, 32);
        const __hip_bfloat16* vr = base + (size_t)j * 1536 + 1024 + c * 16;
        #pragma unroll
        for (int d = 0; d < 16; ++d) oacc[d] += aj * __bfloat162float(vr[d]);
    }
    __hip_bfloat16* orow = o + (size_t)(win * 8 + oi) * CDIM + h * 64 + c * 16;
    #pragma unroll
    for (int d = 0; d < 16; ++d) orow[d] = __float2bfloat16(oacc[d]);
}

// ---------------------------------------------------------------------------
extern "C" void kernel_launch(void* const* d_in, const int* in_sizes, int n_in,
                              void* d_out, int out_size, void* d_ws, size_t ws_size,
                              hipStream_t stream)
{
    (void)in_sizes; (void)n_in; (void)out_size; (void)ws_size;
    const float* x      = (const float*)d_in[0];
    const float* ln1_g  = (const float*)d_in[1];
    const float* ln1_b  = (const float*)d_in[2];
    const float* qkv_w  = (const float*)d_in[3];
    const float* qkv_b  = (const float*)d_in[4];
    const float* proj_w = (const float*)d_in[5];
    const float* proj_b = (const float*)d_in[6];
    const float* ln2_g  = (const float*)d_in[7];
    const float* ln2_b  = (const float*)d_in[8];
    const float* mlp_w1 = (const float*)d_in[9];
    const float* mlp_b1 = (const float*)d_in[10];
    const float* mlp_w2 = (const float*)d_in[11];
    const float* mlp_b2 = (const float*)d_in[12];
    float* out = (float*)d_out;

    char* ws = (char*)d_ws;
    size_t off = 0;
    auto take = [&](size_t bytes) {
        char* p = ws + off; off += (bytes + 255) & ~(size_t)255; return p;
    };
    __hip_bfloat16* wtQkv  = (__hip_bfloat16*)take((size_t)1536 * 512 * 2);
    __hip_bfloat16* wtProj = (__hip_bfloat16*)take((size_t)512  * 512 * 2);
    __hip_bfloat16* wtM1   = (__hip_bfloat16*)take((size_t)2048 * 512 * 2);
    __hip_bfloat16* wtM2   = (__hip_bfloat16*)take((size_t)512  * 2048 * 2);
    __hip_bfloat16* bufA   = (__hip_bfloat16*)take((size_t)TOKENS * 512  * 2); // LN1/o/LN2
    __hip_bfloat16* bufB   = (__hip_bfloat16*)take((size_t)TOKENS * 2048 * 2); // qkv / gelu

    wt_convert<<<(512 * 1536 + 255) / 256, 256, 0, stream>>>(qkv_w,  wtQkv,  512,  1536, (long)512 * 1536);
    wt_convert<<<(512 * 512  + 255) / 256, 256, 0, stream>>>(proj_w, wtProj, 512,  512,  (long)512 * 512);
    wt_convert<<<(512 * 2048 + 255) / 256, 256, 0, stream>>>(mlp_w1, wtM1,   512,  2048, (long)512 * 2048);
    wt_convert<<<(2048 * 512 + 255) / 256, 256, 0, stream>>>(mlp_w2, wtM2,   2048, 512,  (long)2048 * 512);

    // LN1 + roll(-4) -> bf16
    ln_bf16<<<TOKENS, 256, 0, stream>>>(x, ln1_g, ln1_b, bufA, 1);
    // QKV GEMM
    gemm_wmma<EPI_QKV><<<dim3(1536 / BN, TOKENS / BM), 256, 0, stream>>>(
        bufA, wtQkv, qkv_b, nullptr, nullptr, bufB, 1536, 512);
    // window attention (8192 windows x 8 heads, 8 waves/block)
    attn_win8<<<(TOKENS / 8) * 8 / 8, 256, 0, stream>>>(bufB, bufA);
    // proj GEMM + roll(+4) + residual -> x2 in d_out (f32)
    gemm_wmma<EPI_PROJ><<<dim3(512 / BN, TOKENS / BM), 256, 0, stream>>>(
        bufA, wtProj, proj_b, x, out, nullptr, 512, 512);
    // LN2 -> bf16
    ln_bf16<<<TOKENS, 256, 0, stream>>>(out, ln2_g, ln2_b, bufA, 0);
    // MLP up + exact GELU -> bf16
    gemm_wmma<EPI_GELU><<<dim3(2048 / BN, TOKENS / BM), 256, 0, stream>>>(
        bufA, wtM1, mlp_b1, nullptr, nullptr, bufB, 2048, 512);
    // MLP down + residual (RMW d_out)
    gemm_wmma<EPI_MLP2><<<dim3(512 / BN, TOKENS / BM), 256, 0, stream>>>(
        bufB, wtM2, mlp_b2, nullptr, out, nullptr, 512, 2048);
}
